// KANLinear_84516366451195
// MI455X (gfx1250) — compile-verified, matrix operand order checked by reading, not run on hardware
//
#include <hip/hip_runtime.h>
#include <hip/hip_bf16.h>

typedef __attribute__((ext_vector_type(2)))  float   v2f;
typedef __attribute__((ext_vector_type(8)))  float   v8f;
typedef __attribute__((ext_vector_type(8)))  __bf16  v8bf;
typedef __attribute__((ext_vector_type(16))) __bf16  v16bf;

#define B_ROWS 4096
#define IN_F   1024
#define OUT_F  1024
#define NK     8             // spline bases per input feature
#define SLOT   9             // 1 silu/base slot + 8 spline slots
#define KTOT   (IN_F * SLOT) // 9216, divisible by 32

// ---- bf16x2 GEMM tiling (double-buffered async staging) ----
#define BM     128
#define BN     128
#define KS     64            // K-slab (multiple of 32)
#define LDKB   72            // LDS stride (bf16): 16B-aligned, bank-spread
#define NSLAB  (KTOT / KS)   // 144
#define PLANE  (BM * LDKB)   // bf16 elements per plane (9216)
#define BUFSZ  (4 * PLANE)   // bf16 elements per buffer (Ah,Al,Wh,Wl)

// ---- fallback fused-f32 tiling ----
#define ICF    16
#define KSF    (ICF * SLOT)  // 144
#define LDKF   (KSF + 2)     // 146

// Uniform-grid cubic B-spline bases (reference make_grid: h=0.4,
// knots t[j] = -1 + h*(j-3), j=0..11) + SiLU, fully in registers.
__device__ __forceinline__ void silu_and_bsplines(float x, float out9[9]) {
    const float h = 0.4f;
    float d[12];
#pragma unroll
    for (int j = 0; j < 12; ++j)
        d[j] = x - (-1.0f + h * (float)(j - 3));
    float b[11];
#pragma unroll
    for (int j = 0; j < 11; ++j)
        b[j] = (d[j] >= 0.0f && d[j + 1] < 0.0f) ? 1.0f : 0.0f;
#pragma unroll
    for (int k = 1; k <= 3; ++k) {
        const float invkh = 1.0f / (h * (float)k);
#pragma unroll
        for (int j = 0; j + k < 11; ++j)
            b[j] = (d[j] * b[j] - d[j + k + 1] * b[j + 1]) * invkh;
    }
    out9[0] = x / (1.0f + __expf(-x));
#pragma unroll
    for (int j = 0; j < NK; ++j) out9[1 + j] = b[j];
}

// =====================================================================
// Pack kernels: build A = [silu(x) | bases] and W = [base_w | sw*scaler]
// as bf16 hi/lo planes (hi = RNE(v), lo = RNE(v - hi)).
// =====================================================================
__global__ __launch_bounds__(256) void kan_pack_a(
    const float* __restrict__ x,
    __hip_bfloat16* __restrict__ Ahi, __hip_bfloat16* __restrict__ Alo)
{
    const int idx = blockIdx.x * 256 + threadIdx.x;    // b*1024 + i
    const int b = idx >> 10, i = idx & 1023;
    float v9[9];
    silu_and_bsplines(x[idx], v9);
    const size_t base = (size_t)b * KTOT + (size_t)i * SLOT;
#pragma unroll
    for (int q = 0; q < 9; ++q) {
        const float v = v9[q];
        const __hip_bfloat16 h = __float2bfloat16(v);
        Ahi[base + q] = h;
        Alo[base + q] = __float2bfloat16(v - __bfloat162float(h));
    }
}

__global__ __launch_bounds__(256) void kan_pack_w(
    const float* __restrict__ base_w, const float* __restrict__ spline_w,
    const float* __restrict__ scaler,
    __hip_bfloat16* __restrict__ Whi, __hip_bfloat16* __restrict__ Wlo)
{
    const int idx = blockIdx.x * 256 + threadIdx.x;    // o*1024 + i
    const int o = idx >> 10, i = idx & 1023;
    const float sc = scaler[idx];
    const float4* sw = (const float4*)(spline_w + (size_t)idx * NK);
    const float4 s0 = sw[0], s1 = sw[1];
    float v9[9] = { base_w[idx],
                    s0.x * sc, s0.y * sc, s0.z * sc, s0.w * sc,
                    s1.x * sc, s1.y * sc, s1.z * sc, s1.w * sc };
    const size_t base = (size_t)o * KTOT + (size_t)i * SLOT;
#pragma unroll
    for (int q = 0; q < 9; ++q) {
        const float v = v9[q];
        const __hip_bfloat16 h = __float2bfloat16(v);
        Whi[base + q] = h;
        Wlo[base + q] = __float2bfloat16(v - __bfloat162float(h));
    }
}

// =====================================================================
// bf16x2-split GEMM with double-buffered GLOBAL_LOAD_ASYNC_TO_LDS staging
// C(4096x1024) = A(4096x9216) * W(1024x9216)^T
// =====================================================================
__device__ __forceinline__ void async_b128(unsigned int ldsByte,
                                           unsigned int gofByte,
                                           unsigned long long base) {
    // LDS byte address from the vdst VGPR; global address = SGPR base + VGPR offset
    asm volatile("global_load_async_to_lds_b128 %0, %1, %2"
                 :
                 : "v"(ldsByte), "v"(gofByte), "s"(base)
                 : "memory");
}

// Issue one slab's 4-plane staging as async copies (16 x B128 per thread).
__device__ __forceinline__ void stage_slab(
    unsigned int bufOfs /* bf16 elems */, int k0, int tid,
    unsigned long long gAh, unsigned long long gAl,
    unsigned long long gWh, unsigned long long gWl)
{
#pragma unroll
    for (int rep = 0; rep < (BM * KS / 8) / 256; ++rep) {   // 4 reps
        const int q = rep * 256 + tid;
        const int r = q >> 3;              // row 0..127  (KS/8 = 8 chunks/row)
        const int c = (q & 7) << 3;        // bf16 col 0..56
        const unsigned int gof  = (unsigned int)((r * KTOT + k0 + c) * 2);
        const unsigned int lofA = (unsigned int)((bufOfs + r * LDKB + c) * 2);
        async_b128(lofA,                          gof, gAh);
        async_b128(lofA + (unsigned)PLANE * 2,    gof, gAl);
        async_b128(lofA + (unsigned)PLANE * 4,    gof, gWh);
        async_b128(lofA + (unsigned)PLANE * 6,    gof, gWl);
    }
}

__device__ __forceinline__ v16bf frag16(const __hip_bfloat16* p) {
    // ISA 16-bit A/B layout: per lane two 16B chunks at K+0..7 and K+16..23
    const v8bf p0 = *(const v8bf*)p;
    const v8bf p1 = *(const v8bf*)(p + 16);
    return __builtin_shufflevector(p0, p1, 0,1,2,3,4,5,6,7,8,9,10,11,12,13,14,15);
}

__global__ __launch_bounds__(256) void kan_gemm_bf16x2(
    const __hip_bfloat16* __restrict__ Ahi, const __hip_bfloat16* __restrict__ Alo,
    const __hip_bfloat16* __restrict__ Whi, const __hip_bfloat16* __restrict__ Wlo,
    float* __restrict__ out)
{
    extern __shared__ char smem[];
    __hip_bfloat16* sm = (__hip_bfloat16*)smem;   // 2 buffers x 4 planes x BM x LDKB

    const int tid  = threadIdx.x;
    const int lane = tid & 31;
    const int wave = tid >> 5;
    const int m0 = (wave >> 1) * 32;
    const int n0 = (wave & 1) * 64;
    const int rowBase = blockIdx.x * BM;
    const int colBase = blockIdx.y * BN;

    const int fr  = lane & 15;           // fragment row (M for A, N for B)
    const int c8  = (lane >> 4) << 3;    // K sub-offset: lanes 16-31 start at K+8
    const int hi8 = (lane >> 4) << 3;    // C/D upper-half row offset

    // Uniform global bases for this block's row/col panels.
    const unsigned long long gAh = (unsigned long long)(Ahi + (size_t)rowBase * KTOT);
    const unsigned long long gAl = (unsigned long long)(Alo + (size_t)rowBase * KTOT);
    const unsigned long long gWh = (unsigned long long)(Whi + (size_t)colBase * KTOT);
    const unsigned long long gWl = (unsigned long long)(Wlo + (size_t)colBase * KTOT);

    v8f acc[2][4];
#pragma unroll
    for (int tm = 0; tm < 2; ++tm)
#pragma unroll
        for (int tn = 0; tn < 4; ++tn)
            acc[tm][tn] = (v8f){0.f,0.f,0.f,0.f,0.f,0.f,0.f,0.f};

    // Prologue: stage slab 0 into buffer 0.
    stage_slab(0u, 0, tid, gAh, gAl, gWh, gWl);

    for (int s = 0; s < NSLAB; ++s) {
        const unsigned int cur = (s & 1) ? (unsigned)BUFSZ : 0u;
        const unsigned int nxt = (s & 1) ? 0u : (unsigned)BUFSZ;

        // Prefetch next slab into the other buffer (safe: fully consumed,
        // guarded by the trailing barrier of iteration s-1).
        if (s + 1 < NSLAB) {
            stage_slab(nxt, (s + 1) * KS, tid, gAh, gAl, gWh, gWl);
            // async loads complete in order: <=16 outstanding means slab s done
            asm volatile("s_wait_asynccnt 16" ::: "memory");
        } else {
            asm volatile("s_wait_asynccnt 0" ::: "memory");
        }
        __syncthreads();   // slab s visible to all waves

        const __hip_bfloat16* bAh = sm + cur;
        const __hip_bfloat16* bWh = sm + cur + 2 * PLANE;

#pragma unroll
        for (int kb = 0; kb < KS; kb += 32) {
            v16bf ah[2], al[2], bh[4], bl[4];
#pragma unroll
            for (int tm = 0; tm < 2; ++tm) {
                const __hip_bfloat16* p = bAh + (m0 + tm * 16 + fr) * LDKB + kb + c8;
                ah[tm] = frag16(p);
                al[tm] = frag16(p + PLANE);
            }
#pragma unroll
            for (int tn = 0; tn < 4; ++tn) {
                const __hip_bfloat16* p = bWh + (n0 + tn * 16 + fr) * LDKB + kb + c8;
                bh[tn] = frag16(p);
                bl[tn] = frag16(p + PLANE);
            }
#pragma unroll
            for (int tm = 0; tm < 2; ++tm)
#pragma unroll
                for (int tn = 0; tn < 4; ++tn) {
                    acc[tm][tn] = __builtin_amdgcn_wmma_f32_16x16x32_bf16(
                        false, ah[tm], false, bh[tn], (short)0, acc[tm][tn], false, false);
                    acc[tm][tn] = __builtin_amdgcn_wmma_f32_16x16x32_bf16(
                        false, ah[tm], false, bl[tn], (short)0, acc[tm][tn], false, false);
                    acc[tm][tn] = __builtin_amdgcn_wmma_f32_16x16x32_bf16(
                        false, al[tm], false, bh[tn], (short)0, acc[tm][tn], false, false);
                }
        }
        __syncthreads();   // all waves done reading `cur` before it is re-staged
    }

    const int mBase = rowBase + m0 + hi8;
    const int nBase = colBase + n0 + fr;
#pragma unroll
    for (int tm = 0; tm < 2; ++tm)
#pragma unroll
        for (int tn = 0; tn < 4; ++tn)
#pragma unroll
            for (int e = 0; e < 8; ++e)
                out[(size_t)(mBase + tm * 16 + e) * OUT_F + nBase + tn * 16] =
                    acc[tm][tn][e];
}

// =====================================================================
// Fallback: fused f32-WMMA kernel (round-1 design), used if ws too small.
// =====================================================================
__global__ __launch_bounds__(256) void kan_fused_wmma(
    const float* __restrict__ x, const float* __restrict__ base_w,
    const float* __restrict__ spline_w, const float* __restrict__ scaler,
    float* __restrict__ out)
{
    extern __shared__ float lds[];
    float* As = lds;
    float* Ws = lds + BM * LDKF;

    const int tid  = threadIdx.x;
    const int lane = tid & 31;
    const int wave = tid >> 5;
    const int m0 = (wave >> 1) * 32;
    const int n0 = (wave & 1) * 64;
    const int rowBase = blockIdx.x * BM;
    const int colBase = blockIdx.y * BN;
    const int fr = lane & 15;
    const int kk = (lane >> 4) << 1;
    const int hi8 = (lane >> 4) << 3;

    v8f acc[2][4];
#pragma unroll
    for (int tm = 0; tm < 2; ++tm)
#pragma unroll
        for (int tn = 0; tn < 4; ++tn)
            acc[tm][tn] = (v8f){0.f,0.f,0.f,0.f,0.f,0.f,0.f,0.f};

    for (int s = 0; s < IN_F / ICF; ++s) {
        const int i0 = s * ICF;
#pragma unroll
        for (int rep = 0; rep < (BM * ICF) / 256; ++rep) {
            const int p = rep * 256 + tid;
            const int m = p >> 4, f = p & 15;
            float v9[9];
            silu_and_bsplines(x[(size_t)(rowBase + m) * IN_F + i0 + f], v9);
            float* dst = &As[m * LDKF + f * SLOT];
#pragma unroll
            for (int q = 0; q < 9; ++q) dst[q] = v9[q];
        }
#pragma unroll
        for (int rep = 0; rep < (BN * ICF) / 256; ++rep) {
            const int p = rep * 256 + tid;
            const int n = p >> 4, f = p & 15;
            const size_t g = (size_t)(colBase + n) * IN_F + i0 + f;
            const float bw = base_w[g], sc = scaler[g];
            const float4* sw = (const float4*)(spline_w + g * NK);
            const float4 s0 = sw[0], s1 = sw[1];
            float* dst = &Ws[n * LDKF + f * SLOT];
            dst[0] = bw;
            dst[1] = s0.x * sc; dst[2] = s0.y * sc; dst[3] = s0.z * sc; dst[4] = s0.w * sc;
            dst[5] = s1.x * sc; dst[6] = s1.y * sc; dst[7] = s1.z * sc; dst[8] = s1.w * sc;
        }
        __syncthreads();
        for (int kb = 0; kb < KSF; kb += 4) {
            v2f a[2], b[4];
#pragma unroll
            for (int tm = 0; tm < 2; ++tm)
                a[tm] = *(const v2f*)&As[(m0 + tm * 16 + fr) * LDKF + kb + kk];
#pragma unroll
            for (int tn = 0; tn < 4; ++tn)
                b[tn] = *(const v2f*)&Ws[(n0 + tn * 16 + fr) * LDKF + kb + kk];
#pragma unroll
            for (int tm = 0; tm < 2; ++tm)
#pragma unroll
                for (int tn = 0; tn < 4; ++tn)
                    acc[tm][tn] = __builtin_amdgcn_wmma_f32_16x16x4_f32(
                        false, a[tm], false, b[tn], (short)0, acc[tm][tn], false, false);
        }
        __syncthreads();
    }

    const int mBase = rowBase + m0 + hi8;
    const int nBase = colBase + n0 + fr;
#pragma unroll
    for (int tm = 0; tm < 2; ++tm)
#pragma unroll
        for (int tn = 0; tn < 4; ++tn)
#pragma unroll
            for (int e = 0; e < 8; ++e)
                out[(size_t)(mBase + tm * 16 + e) * OUT_F + nBase + tn * 16] =
                    acc[tm][tn][e];
}

// =====================================================================
extern "C" void kernel_launch(void* const* d_in, const int* in_sizes, int n_in,
                              void* d_out, int out_size, void* d_ws, size_t ws_size,
                              hipStream_t stream) {
    const float* x  = (const float*)d_in[0];
    const float* bw = (const float*)d_in[1];
    const float* sw = (const float*)d_in[2];
    const float* sc = (const float*)d_in[3];
    // d_in[4] = grid: uniform knots reconstructed from constants in-kernel.
    float* out = (float*)d_out;

    const size_t nA = (size_t)B_ROWS * KTOT;
    const size_t nW = (size_t)OUT_F * KTOT;
    const size_t need = (2 * nA + 2 * nW) * sizeof(__hip_bfloat16); // ~189 MB

    if (ws_size >= need) {
        __hip_bfloat16* Ahi = (__hip_bfloat16*)d_ws;
        __hip_bfloat16* Alo = Ahi + nA;
        __hip_bfloat16* Whi = Alo + nA;
        __hip_bfloat16* Wlo = Whi + nW;
        kan_pack_a<<<(B_ROWS * IN_F) / 256, 256, 0, stream>>>(x, Ahi, Alo);
        kan_pack_w<<<(OUT_F * IN_F) / 256, 256, 0, stream>>>(bw, sw, sc, Whi, Wlo);
        const size_t shg = (size_t)2 * BUFSZ * sizeof(__hip_bfloat16); // 144 KB
        hipFuncSetAttribute((const void*)kan_gemm_bf16x2,
                            hipFuncAttributeMaxDynamicSharedMemorySize, (int)shg);
        dim3 g(B_ROWS / BM, OUT_F / BN);
        kan_gemm_bf16x2<<<g, 256, shg, stream>>>(Ahi, Alo, Whi, Wlo, out);
    } else {
        const size_t shf = (size_t)(BM + BN) * LDKF * sizeof(float); // ~146 KB
        hipFuncSetAttribute((const void*)kan_fused_wmma,
                            hipFuncAttributeMaxDynamicSharedMemorySize, (int)shf);
        dim3 g(B_ROWS / BM, OUT_F / BN);
        kan_fused_wmma<<<g, 256, shf, stream>>>(x, bw, sw, sc, out);
    }
}
